// KANConvNetVanilla_17995912970409
// MI455X (gfx1250) — compile-verified, matrix operand order checked by reading, not run on hardware
//
#include <hip/hip_runtime.h>

typedef __attribute__((ext_vector_type(16))) _Float16 v16h;
typedef __attribute__((ext_vector_type(8)))  float    v8f;

#define KC    288   // K-chunk: 32 patch elements * 9 features, = 9 WMMA k-steps
#define ROWP  296   // LDS row stride in halves (148 dwords -> bank-conflict free)
#define TILE_M 64   // samples per block (4 waves * 16)

// ---------------------------------------------------------------------------
// Uniform cubic B-spline (grid_size=5, order=3, range [-1,1], h=0.4) + silu.
// Knots t_j = -2.2 + 0.4*j, j=0..11.  Produces f[0]=silu(x), f[1..8]=B_k(x).
// ---------------------------------------------------------------------------
__device__ __forceinline__ void kan_features(float x, float f[9]) {
  f[0] = x / (1.f + __expf(-x));                 // silu
  const float h  = 0.4f;
  const float t0 = -1.f - 3.f * h;
  float b0[11];
#pragma unroll
  for (int j = 0; j < 11; ++j) {
    float tj = t0 + h * j;
    b0[j] = (x >= tj && x < tj + h) ? 1.f : 0.f;
  }
  float b1[10];
#pragma unroll
  for (int j = 0; j < 10; ++j) {
    float tj = t0 + h * j;
    b1[j] = (x - tj) * (1.f / h) * b0[j] + ((tj + 2.f * h) - x) * (1.f / h) * b0[j + 1];
  }
  float b2[9];
#pragma unroll
  for (int j = 0; j < 9; ++j) {
    float tj = t0 + h * j;
    b2[j] = (x - tj) * (1.f / (2.f * h)) * b1[j] + ((tj + 3.f * h) - x) * (1.f / (2.f * h)) * b1[j + 1];
  }
#pragma unroll
  for (int j = 0; j < 8; ++j) {
    float tj = t0 + h * j;
    f[1 + j] = (x - tj) * (1.f / (3.f * h)) * b2[j] + ((tj + 4.f * h) - x) * (1.f / (3.f * h)) * b2[j + 1];
  }
}

// ---------------------------------------------------------------------------
// Repack KAN weights into augmented f16 GEMM layout:
//   Wp[o][chunk*288 + (i%32)*9 + j],  j=0 -> bw[o,i], j=1..8 -> sw[o,i,j-1]*sc[o,i]
// Rows o >= outC and columns i >= inF are zero-filled.
// ---------------------------------------------------------------------------
__global__ void prep_weights(const float* __restrict__ bw, const float* __restrict__ sw,
                             const float* __restrict__ sc, _Float16* __restrict__ Wp,
                             int outC, int outPad, int inF, int inPad, int Kstride)
{
  int idx = blockIdx.x * blockDim.x + threadIdx.x;
  int total = outPad * inPad;
  if (idx >= total) return;
  int i = idx % inPad;
  int o = idx / inPad;
  _Float16* dst = Wp + (size_t)o * Kstride + (size_t)(i >> 5) * KC + (i & 31) * 9;
  if (o < outC && i < inF) {
    int oi = o * inF + i;
    float s = sc[oi];
    dst[0] = (_Float16)bw[oi];
#pragma unroll
    for (int j = 0; j < 8; ++j) dst[1 + j] = (_Float16)(sw[oi * 8 + j] * s);
  } else {
#pragma unroll
    for (int j = 0; j < 9; ++j) dst[j] = (_Float16)0.f;
  }
}

// ---------------------------------------------------------------------------
// Fused KAN layer: on-the-fly (im2col + spline-feature) -> LDS -> WMMA GEMM.
// mode==1: conv (X is NCHW, samples are output pixels); mode==0: plain linear.
// Block = 128 threads (4 waves). Block tile: 64 samples x 16 out channels.
// Per wave: one 16x16 D tile via v_wmma_f32_16x16x32_f16, K chunked by 288.
// ---------------------------------------------------------------------------
__global__ void __launch_bounds__(128)
kan_layer(const float* __restrict__ X, const _Float16* __restrict__ Wp,
          float* __restrict__ Y,
          int mode, int N, int inF, int nChunk, int Kstride,
          int outC, int doRelu,
          int Cin, int H, int W, int kk, int st, int pd, int Ho, int Wo)
{
  __shared__ _Float16 feat[TILE_M][ROWP];  // 64 x 296 halves = 37.9 KB
  __shared__ _Float16 wt[16][ROWP];        //  9.5 KB

  const int tid  = threadIdx.x;
  const int lane = tid & 31;
  const int wave = tid >> 5;
  const int s0   = blockIdx.x * TILE_M;
  const int ocb  = blockIdx.y * 16;

  const int g  = lane >> 4;       // K-half select within fragment layout
  const int nn = lane & 15;       // out-channel column (B/D), sample row (A)
  const int ml = (wave << 4) + nn;

  v8f acc = {0.f, 0.f, 0.f, 0.f, 0.f, 0.f, 0.f, 0.f};

  for (int c = 0; c < nChunk; ++c) {
    const int iBase = c << 5;     // first patch element of this chunk
    __syncthreads();              // previous chunk's WMMA reads done

    // ---- phase 1: 64 samples x 32 patch elems -> 9 f16 features each ----
#pragma unroll 4
    for (int r = 0; r < 16; ++r) {
      int p  = (r << 7) + tid;    // 0..2047
      int m  = p >> 5;
      int il = p & 31;
      int i  = iBase + il;
      int s  = s0 + m;
      float f[9];
      if (s < N && i < inF) {
        float x = 0.f;            // zero-padding still gets full spline features
        if (mode == 1) {
          int wo = s % Wo; int t = s / Wo; int ho = t % Ho; int bb = t / Ho;
          int ksq = kk * kk;
          int ci = i / ksq; int rem = i - ci * ksq;
          int dy = rem / kk; int dx = rem - dy * kk;
          int iy = ho * st - pd + dy; int ix = wo * st - pd + dx;
          if (iy >= 0 && iy < H && ix >= 0 && ix < W)
            x = X[(((size_t)bb * Cin + ci) * H + iy) * W + ix];
        } else {
          x = X[(size_t)s * inF + i];
        }
        kan_features(x, f);
      } else {
#pragma unroll
        for (int j = 0; j < 9; ++j) f[j] = 0.f;
      }
      _Float16* dst = &feat[m][il * 9];
#pragma unroll
      for (int j = 0; j < 9; ++j) dst[j] = (_Float16)f[j];
    }

    // ---- weights chunk -> LDS (16 rows x 288 halves) ----
    const _Float16* wsrc = Wp + (size_t)ocb * Kstride + (size_t)c * KC;
    for (int q = tid; q < 16 * KC; q += 128) {
      int row = q / KC, col = q - row * KC;
      wt[row][col] = wsrc[(size_t)row * Kstride + col];
    }
    if (c + 1 < nChunk && tid < 16)
      __builtin_prefetch(wsrc + (size_t)tid * Kstride + KC, 0, 1);  // global_prefetch_b8
    __syncthreads();

    // ---- phase 2: 9 WMMA k-steps over the 288-wide chunk ----
#pragma unroll
    for (int kb = 0; kb < KC; kb += 32) {
      v16h a, bm;
#pragma unroll
      for (int r = 0; r < 8; ++r) {
        // 16-bit fragment layout: K = 2*(r%4) + h + 8*g + 16*(r/4)
        int koff = kb + ((r & 3) << 1) + (g << 3) + ((r >> 2) << 4);
        a[2 * r]      = feat[ml][koff];
        a[2 * r + 1]  = feat[ml][koff + 1];
        bm[2 * r]     = wt[nn][koff];
        bm[2 * r + 1] = wt[nn][koff + 1];
      }
      acc = __builtin_amdgcn_wmma_f32_16x16x32_f16(false, a, false, bm,
                                                   (short)0, acc, false, false);
    }
  }

  // ---- store: D element (M = v + 8g, N = lane%16) ----
#pragma unroll
  for (int v = 0; v < 8; ++v) {
    int s  = s0 + (wave << 4) + (g << 3) + v;
    int oc = ocb + nn;
    if (s < N && oc < outC) {
      float val = acc[v];
      if (doRelu) val = fmaxf(val, 0.f);
      if (mode == 1) {
        int wo = s % Wo; int t = s / Wo; int ho = t % Ho; int bb = t / Ho;
        Y[(((size_t)bb * outC + oc) * Ho + ho) * Wo + wo] = val;   // NCHW
      } else {
        Y[(size_t)s * outC + oc] = val;
      }
    }
  }
}

// ---------------------------------------------------------------------------
// 2x2 max pool, NCHW. (H, W) are OUTPUT dims.
// ---------------------------------------------------------------------------
__global__ void maxpool2x2(const float* __restrict__ X, float* __restrict__ Y,
                           int BC, int H, int W)
{
  int idx = blockIdx.x * blockDim.x + threadIdx.x;
  int total = BC * H * W;
  if (idx >= total) return;
  int w = idx % W; int t = idx / W; int hh = t % H; int bc = t / H;
  const float* p = X + ((size_t)bc * (2 * H) + 2 * hh) * (2 * W) + 2 * w;
  float v = fmaxf(fmaxf(p[0], p[1]), fmaxf(p[2 * W], p[2 * W + 1]));
  Y[idx] = v;
}

// Global average pool over 7x7 per (b, c).
__global__ void mean49(const float* __restrict__ X, float* __restrict__ Y)
{
  int idx = blockIdx.x * blockDim.x + threadIdx.x;
  if (idx >= 8 * 64) return;
  const float* p = X + (size_t)idx * 49;
  float s = 0.f;
#pragma unroll
  for (int i = 0; i < 49; ++i) s += p[i];
  Y[idx] = s * (1.f / 49.f);
}

// ---------------------------------------------------------------------------
extern "C" void kernel_launch(void* const* d_in, const int* in_sizes, int n_in,
                              void* d_out, int out_size, void* d_ws, size_t ws_size,
                              hipStream_t stream)
{
  const float* x   = (const float*)d_in[0];
  const float* bw0 = (const float*)d_in[1];
  const float* sw0 = (const float*)d_in[2];
  const float* sc0 = (const float*)d_in[3];
  const float* bw1 = (const float*)d_in[4];
  const float* sw1 = (const float*)d_in[5];
  const float* sc1 = (const float*)d_in[6];
  const float* bw2 = (const float*)d_in[7];
  const float* sw2 = (const float*)d_in[8];
  const float* sc2 = (const float*)d_in[9];
  const float* bwc = (const float*)d_in[10];
  const float* swc = (const float*)d_in[11];
  const float* scc = (const float*)d_in[12];

  // ---- carve workspace ----
  char* p = (char*)d_ws;
  auto carve = [&](size_t bytes) -> char* {
    char* r = p; p += (bytes + 255) & ~(size_t)255; return r;
  };
  _Float16* Wp0 = (_Float16*)carve((size_t)16  * 1440 * 2);
  _Float16* Wp1 = (_Float16*)carve((size_t)32  * 1440 * 2);
  _Float16* Wp2 = (_Float16*)carve((size_t)64  * 2592 * 2);
  _Float16* Wpc = (_Float16*)carve((size_t)208 * 576  * 2);
  float* act0  = (float*)carve((size_t)100352 * 16 * 4);  // 8x16x112x112
  float* act0p = (float*)carve((size_t)8 * 16 * 56 * 56 * 4);
  float* act1  = (float*)carve((size_t)6272 * 32 * 4);    // 8x32x28x28
  float* act1p = (float*)carve((size_t)8 * 32 * 14 * 14 * 4);
  float* act2  = (float*)carve((size_t)392 * 64 * 4);     // 8x64x7x7
  float* pooled = (float*)carve((size_t)512 * 4);         // 8x64

  // ---- repack weights (augmented f16 layout) ----
  prep_weights<<<dim3((16  * 160 + 255) / 256), dim3(256), 0, stream>>>(bw0, sw0, sc0, Wp0, 16,  16,  147, 160, 1440);
  prep_weights<<<dim3((32  * 160 + 255) / 256), dim3(256), 0, stream>>>(bw1, sw1, sc1, Wp1, 32,  32,  144, 160, 1440);
  prep_weights<<<dim3((64  * 288 + 255) / 256), dim3(256), 0, stream>>>(bw2, sw2, sc2, Wp2, 64,  64,  288, 288, 2592);
  prep_weights<<<dim3((208 * 64  + 255) / 256), dim3(256), 0, stream>>>(bwc, swc, scc, Wpc, 200, 208, 64,  64,  576);

  const dim3 blk(128);

  // conv0: 3->16, k=7, s=2, p=3, 224 -> 112; N=100352, K=147*9 (5 chunks)
  kan_layer<<<dim3(100352 / TILE_M, 1), blk, 0, stream>>>(
      x, Wp0, act0, 1, 100352, 147, 5, 1440, 16, 1,
      3, 224, 224, 7, 2, 3, 112, 112);
  maxpool2x2<<<dim3((8 * 16 * 56 * 56 + 255) / 256), dim3(256), 0, stream>>>(act0, act0p, 8 * 16, 56, 56);

  // conv1: 16->32, k=3, s=2, p=1, 56 -> 28; N=6272, K=144*9 (5 chunks)
  kan_layer<<<dim3(6272 / TILE_M, 2), blk, 0, stream>>>(
      act0p, Wp1, act1, 1, 6272, 144, 5, 1440, 32, 1,
      16, 56, 56, 3, 2, 1, 28, 28);
  maxpool2x2<<<dim3((8 * 32 * 14 * 14 + 255) / 256), dim3(256), 0, stream>>>(act1, act1p, 8 * 32, 14, 14);

  // conv2: 32->64, k=3, s=2, p=1, 14 -> 7; N=392, K=288*9 (9 chunks)
  kan_layer<<<dim3((392 + TILE_M - 1) / TILE_M, 4), blk, 0, stream>>>(
      act1p, Wp2, act2, 1, 392, 288, 9, 2592, 64, 1,
      32, 14, 14, 3, 2, 1, 7, 7);

  mean49<<<dim3(2), dim3(256), 0, stream>>>(act2, pooled);

  // classifier: 64 -> 200 (13 tiles of 16 out), no relu, linear mode
  kan_layer<<<dim3(1, 13), blk, 0, stream>>>(
      pooled, Wpc, (float*)d_out, 0, 8, 64, 2, 576, 200, 0,
      1, 1, 1, 1, 1, 1, 1, 1);
}
